// RotaryMultiheadAttention_70042326663542
// MI455X (gfx1250) — matrix operand order repaired and checked by reading, not compiled, not run on hardware
//
#include <hip/hip_runtime.h>
#include <hip/hip_bf16.h>

// ---------------------------------------------------------------------------
// Types for CDNA5 WMMA (wave32, 16x16x32 f16 -> f32)
// ---------------------------------------------------------------------------
typedef __attribute__((ext_vector_type(16))) _Float16 v16h;
typedef __attribute__((ext_vector_type(8)))  _Float16 v8h;
typedef __attribute__((ext_vector_type(8)))  float    v8f;

#define B_SZ   4
#define T_SZ   2048
#define D_SZ   1024
#define H_SZ   16
#define HD_SZ  64
#define NROWS  (B_SZ * T_SZ)          // 8192
#define LDPAD  40                     // LDS row stride in halves (padded vs 32)
#define KPAD   72                     // LDS row stride for 64-wide K rows

// ---------------------------------------------------------------------------
// CDNA5 async global->LDS copy (ASYNCcnt-tracked), 16 bytes per lane.
//   LDS address from the vdst VGPR;  global addr from VGPR pair.
// ---------------------------------------------------------------------------
__device__ __forceinline__ void async_ld_b128(void* lds, const void* g) {
  unsigned l = (unsigned)(unsigned long long)lds;   // low 32 bits = LDS offset
  asm volatile("global_load_async_to_lds_b128 %0, %1, off"
               :: "v"(l), "v"(g) : "memory");
}
__device__ __forceinline__ void wait_async0() {
  asm volatile("s_wait_asynccnt 0" ::: "memory");
}

// A-fragment (16x32, 16-bit): lane L holds row L%16; VGPR0-3 hold K = kb..kb+7,
// VGPR4-7 hold K = 16+kb..16+kb+7 with kb = (L/16)*8.  Source must be row-major
// with K contiguous.
__device__ __forceinline__ v16h load_fragA(const _Float16* p, int ld) {
  const int lane = threadIdx.x & 31;
  const int row  = lane & 15;
  const int kb   = (lane >> 4) << 3;           // 0 or 8
  const _Float16* r0 = p + row * ld + kb;
  v8h lo = *(const v8h*)(r0);
  v8h hi = *(const v8h*)(r0 + 16);
  v16h out;
#pragma unroll
  for (int i = 0; i < 8; ++i) { out[i] = lo[i]; out[i + 8] = hi[i]; }
  return out;
}

// B-fragment (32x16, 16-bit): lane L holds column L%16; VGPR0-3 hold K = kb..kb+7,
// VGPR4-7 hold K = 16+kb..16+kb+7 with kb = (L/16)*8.  Source is
// row-major [N][K] (each output column's K run contiguous).
__device__ __forceinline__ v16h load_fragB(const _Float16* p, int ld) {
  const int lane = threadIdx.x & 31;
  const int col  = lane & 15;
  const int kb   = (lane >> 4) << 3;           // 0 or 8
  const _Float16* r0 = p + col * ld + kb;
  v8h lo = *(const v8h*)(r0);
  v8h hi = *(const v8h*)(r0 + 16);
  v16h out;
#pragma unroll
  for (int i = 0; i < 8; ++i) { out[i] = lo[i]; out[i + 8] = hi[i]; }
  return out;
}

__device__ __forceinline__ v8f wmma16(v16h a, v16h b, v8f c) {
  return __builtin_amdgcn_wmma_f32_16x16x32_f16(false, a, false, b,
                                                (short)0, c, false, false);
}

// ---------------------------------------------------------------------------
// Kernel 0: f32 -> f16 conversion (weights), 4 elements / thread
// ---------------------------------------------------------------------------
__global__ void cvt_f32_f16(const float* __restrict__ s,
                            _Float16* __restrict__ d, int n) {
  int i = (blockIdx.x * blockDim.x + threadIdx.x) * 4;
  if (i + 3 < n) {
    float4 f = *(const float4*)(s + i);
    d[i + 0] = (_Float16)f.x;
    d[i + 1] = (_Float16)f.y;
    d[i + 2] = (_Float16)f.z;
    d[i + 3] = (_Float16)f.w;
  } else {
    for (; i < n; ++i) d[i] = (_Float16)s[i];
  }
}

// ---------------------------------------------------------------------------
// Kernel 1: QKV = query @ in_proj_w^T + b, fused RoPE + scale + head scatter.
//   query: f32 [8192,1024]    in_proj_w: f16 [3072,1024]   bias: f32 [3072]
//   Qo,Ko: f16 [b,h,t,64]  (Q pre-scaled by 1/8)   Vt: f16 [b,h,64,t]
// Block = 256 thr (8 waves), tile 128x128, K-step 32.
// B tile copied with async global->LDS; A tile converted f32->f16 in VGPRs.
// ---------------------------------------------------------------------------
__global__ __launch_bounds__(256, 1)
void gemm_qkv_rope(const float* __restrict__ A,
                   const _Float16* __restrict__ W,
                   const float* __restrict__ bias,
                   _Float16* __restrict__ Qo,
                   _Float16* __restrict__ Ko,
                   _Float16* __restrict__ Vt) {
  __shared__ _Float16 As[128 * LDPAD];
  __shared__ _Float16 Bs[128 * LDPAD];
  const int tiN = blockIdx.x;            // 0..23
  const int tiM = blockIdx.y;            // 0..63
  const int tid = threadIdx.x;
  const int wave = tid >> 5, lane = tid & 31;
  const int wrg = (wave >> 1) * 32;      // wave row group: 0,32,64,96
  const int wcg = (wave & 1) * 64;       // wave col group: 0,64

  v8f acc[2][4] = {};

  for (int k0 = 0; k0 < D_SZ; k0 += 32) {
#pragma unroll
    for (int it = 0; it < 2; ++it) {
      int idx = it * 256 + tid;          // 0..511
      int r = idx >> 2;                  // 0..127
      int c = (idx & 3) * 8;             // 0,8,16,24
      // B: async byte-copy global -> LDS (already f16)
      async_ld_b128(&Bs[r * LDPAD + c],
                    &W[(size_t)(tiN * 128 + r) * D_SZ + k0 + c]);
      // A: f32 -> f16 on the fly
      const float* ap = A + (size_t)(tiM * 128 + r) * D_SZ + k0 + c;
      float4 f0 = *(const float4*)ap;
      float4 f1 = *(const float4*)(ap + 4);
      v8h ha;
      ha[0] = (_Float16)f0.x; ha[1] = (_Float16)f0.y;
      ha[2] = (_Float16)f0.z; ha[3] = (_Float16)f0.w;
      ha[4] = (_Float16)f1.x; ha[5] = (_Float16)f1.y;
      ha[6] = (_Float16)f1.z; ha[7] = (_Float16)f1.w;
      *(v8h*)&As[r * LDPAD + c] = ha;
    }
    if (k0 + 32 < D_SZ)
      __builtin_prefetch(&A[(size_t)(tiM * 128 + (tid >> 1)) * D_SZ + k0 + 32], 0, 3);
    wait_async0();
    __syncthreads();

    v16h af[2], bf[4];
#pragma unroll
    for (int i = 0; i < 2; ++i) af[i] = load_fragA(&As[(wrg + 16 * i) * LDPAD], LDPAD);
#pragma unroll
    for (int j = 0; j < 4; ++j) bf[j] = load_fragB(&Bs[(wcg + 16 * j) * LDPAD], LDPAD);
#pragma unroll
    for (int i = 0; i < 2; ++i)
#pragma unroll
      for (int j = 0; j < 4; ++j)
        acc[i][j] = wmma16(af[i], bf[j], acc[i][j]);
    __syncthreads();
  }

  // Epilogue: bias + RoPE + scatter.  One 128-col tile spans exactly 2 heads
  // and lies in a single q/k/v section.
  const int sec = (tiN * 128) >> 10;     // 0=Q,1=K,2=V (uniform per block)
#pragma unroll
  for (int i = 0; i < 2; ++i) {
#pragma unroll
    for (int j = 0; j < 4; ++j) {
      const int colLocal = wcg + 16 * j + (lane & 15);
      const int N  = tiN * 128 + colLocal;
      const int n1 = N & 1023;
      const int hh = n1 >> 6;
      const int dd = n1 & 63;
      const float bv = bias[N];
      const float invf = __expf(-(float)(dd & ~1) * 0.14391156f); // ln(1e4)/64
#pragma unroll
      for (int r = 0; r < 8; ++r) {
        const int rowLocal = wrg + 16 * i + 8 * (lane >> 4) + r;
        const int gRow = tiM * 128 + rowLocal;
        const int bb = gRow >> 11;
        const int tt = gRow & 2047;
        float val = acc[i][j][r] + bv;
        if (sec < 2) {
          float p = __shfl_xor(val, 1);        // partner column of the pair
          float th = (float)tt * invf;
          float cs = __cosf(th), sn = __sinf(th);
          val = val * cs + ((dd & 1) ? p * sn : -p * sn);
        }
        if (sec == 0) {
          Qo[(((size_t)(bb * H_SZ + hh) * T_SZ) + tt) * HD_SZ + dd] =
              (_Float16)(val * 0.125f);        // fold 1/sqrt(64)
        } else if (sec == 1) {
          Ko[(((size_t)(bb * H_SZ + hh) * T_SZ) + tt) * HD_SZ + dd] =
              (_Float16)val;
        } else {
          Vt[(((size_t)(bb * H_SZ + hh) * HD_SZ) + dd) * T_SZ + tt] =
              (_Float16)val;
        }
      }
    }
  }
}

// ---------------------------------------------------------------------------
// Kernel 2: flash attention.  grid.x = b*h (64), grid.y = query tile (8).
// 8 waves / block; each wave owns 32 query rows.  K/V 32-key tiles are staged
// once per block into LDS via async copies and shared by all waves.
// ---------------------------------------------------------------------------
__global__ __launch_bounds__(256, 1)
void attn_kernel(const _Float16* __restrict__ Qf,
                 const _Float16* __restrict__ Kf,
                 const _Float16* __restrict__ Vt,
                 _Float16* __restrict__ Ob) {
  __shared__ _Float16 Ks[32 * KPAD];           // keys x head_dim
  __shared__ _Float16 Vs[64 * LDPAD];          // head_dim x keys
  __shared__ _Float16 Pbuf[8 * 32 * LDPAD];    // per-wave P tile (32x32)
  const int bh = blockIdx.x;
  const int bb = bh >> 4, hh = bh & 15;
  const int tid = threadIdx.x;
  const int wave = tid >> 5, lane = tid & 31;
  const int q0 = blockIdx.y * 256 + wave * 32;

  const _Float16* Qp = Qf + ((size_t)bh * T_SZ + q0) * HD_SZ;
  const _Float16* Kp = Kf + (size_t)bh * T_SZ * HD_SZ;
  const _Float16* Vp = Vt + (size_t)bh * HD_SZ * T_SZ;
  _Float16* Pw = &Pbuf[wave * 32 * LDPAD];

  // Q fragments: rows 16*i, K chunks of 32 over head_dim
  v16h qfr[2][2];
#pragma unroll
  for (int i = 0; i < 2; ++i)
#pragma unroll
    for (int c = 0; c < 2; ++c)
      qfr[i][c] = load_fragA(Qp + (size_t)(16 * i) * HD_SZ + 32 * c, HD_SZ);

  v8f o[2][4] = {};
  float m[2][8], l[2][8];
#pragma unroll
  for (int i = 0; i < 2; ++i)
#pragma unroll
    for (int r = 0; r < 8; ++r) { m[i][r] = -1e30f; l[i][r] = 0.0f; }

  // cooperative staging indices (one b128 per thread per tile)
  const int kr = tid >> 3, kc = (tid & 7) * 8;   // K: key row, dim chunk
  const int vr = tid >> 2, vc = (tid & 3) * 8;   // V: dim row, key chunk

  for (int kt = 0; kt < T_SZ; kt += 32) {
    // stage K (32x64) and V (64x32) tiles into LDS, shared by all 8 waves
    async_ld_b128(&Ks[kr * KPAD + kc], Kp + (size_t)(kt + kr) * HD_SZ + kc);
    async_ld_b128(&Vs[vr * LDPAD + vc], Vp + (size_t)vr * T_SZ + kt + vc);
    wait_async0();
    __syncthreads();

    // S = Q @ K^T  (32 queries x 32 keys), f32 accum
    v8f s[2][2] = {};
#pragma unroll
    for (int j = 0; j < 2; ++j)
#pragma unroll
      for (int c = 0; c < 2; ++c) {
        v16h kfr = load_fragB(&Ks[(16 * j) * KPAD + 32 * c], KPAD);
        s[0][j] = wmma16(qfr[0][c], kfr, s[0][j]);
        s[1][j] = wmma16(qfr[1][c], kfr, s[1][j]);
      }

    // online softmax (rows live in VGPR index, cols across 16-lane halves)
#pragma unroll
    for (int i = 0; i < 2; ++i) {
      float mn[8], corr[8], rs[8];
#pragma unroll
      for (int r = 0; r < 8; ++r) mn[r] = fmaxf(s[i][0][r], s[i][1][r]);
#pragma unroll
      for (int mk = 1; mk < 16; mk <<= 1)
#pragma unroll
        for (int r = 0; r < 8; ++r) mn[r] = fmaxf(mn[r], __shfl_xor(mn[r], mk));
#pragma unroll
      for (int r = 0; r < 8; ++r) {
        float mnew = fmaxf(mn[r], m[i][r]);
        corr[r] = __expf(m[i][r] - mnew);
        m[i][r] = mnew;
      }
#pragma unroll
      for (int r = 0; r < 8; ++r) {
        float p0 = __expf(s[i][0][r] - m[i][r]);
        float p1 = __expf(s[i][1][r] - m[i][r]);
        s[i][0][r] = p0; s[i][1][r] = p1;
        rs[r] = p0 + p1;
      }
#pragma unroll
      for (int mk = 1; mk < 16; mk <<= 1)
#pragma unroll
        for (int r = 0; r < 8; ++r) rs[r] += __shfl_xor(rs[r], mk);
#pragma unroll
      for (int r = 0; r < 8; ++r) l[i][r] = l[i][r] * corr[r] + rs[r];
#pragma unroll
      for (int nj = 0; nj < 4; ++nj)
#pragma unroll
        for (int r = 0; r < 8; ++r) o[i][nj][r] *= corr[r];

      // P (C-layout) -> LDS row-major so it can re-enter as an A fragment
#pragma unroll
      for (int j = 0; j < 2; ++j)
#pragma unroll
        for (int r = 0; r < 8; ++r)
          Pw[(16 * i + 8 * (lane >> 4) + r) * LDPAD + 16 * j + (lane & 15)] =
              (_Float16)s[i][j][r];
    }

    // O += P @ V  (per-wave P buffer: same-wave DS ordering suffices)
    v16h vfr[4];
#pragma unroll
    for (int nj = 0; nj < 4; ++nj)
      vfr[nj] = load_fragB(&Vs[(16 * nj) * LDPAD], LDPAD);
#pragma unroll
    for (int i = 0; i < 2; ++i) {
      v16h pfr = load_fragA(Pw + 16 * i * LDPAD, LDPAD);
#pragma unroll
      for (int nj = 0; nj < 4; ++nj)
        o[i][nj] = wmma16(pfr, vfr[nj], o[i][nj]);
    }
    __syncthreads();   // all waves done with Ks/Vs before next overwrite
  }

  // normalize + store to [b, t, h*64+d] (row-major for the out-proj GEMM)
#pragma unroll
  for (int i = 0; i < 2; ++i)
#pragma unroll
    for (int nj = 0; nj < 4; ++nj)
#pragma unroll
      for (int r = 0; r < 8; ++r) {
        const int rowLocal = 16 * i + 8 * (lane >> 4) + r;
        const int tt = q0 + rowLocal;
        const int dd = 16 * nj + (lane & 15);
        Ob[((size_t)(bb * T_SZ + tt)) * D_SZ + hh * HD_SZ + dd] =
            (_Float16)(o[i][nj][r] / l[i][r]);
      }
}

// ---------------------------------------------------------------------------
// Kernel 3: out = attn @ out_proj_w^T + b.  f16 inputs, f32 output.
// Both tiles copied global->LDS with async b128.
// ---------------------------------------------------------------------------
__global__ __launch_bounds__(256, 1)
void gemm_out(const _Float16* __restrict__ A,
              const _Float16* __restrict__ W,
              const float* __restrict__ bias,
              float* __restrict__ out) {
  __shared__ _Float16 As[128 * LDPAD];
  __shared__ _Float16 Bs[128 * LDPAD];
  const int tiN = blockIdx.x;            // 0..7
  const int tiM = blockIdx.y;            // 0..63
  const int tid = threadIdx.x;
  const int wave = tid >> 5, lane = tid & 31;
  const int wrg = (wave >> 1) * 32;
  const int wcg = (wave & 1) * 64;

  v8f acc[2][4] = {};

  for (int k0 = 0; k0 < D_SZ; k0 += 32) {
#pragma unroll
    for (int it = 0; it < 2; ++it) {
      int idx = it * 256 + tid;
      int r = idx >> 2;
      int c = (idx & 3) * 8;
      async_ld_b128(&As[r * LDPAD + c],
                    &A[(size_t)(tiM * 128 + r) * D_SZ + k0 + c]);
      async_ld_b128(&Bs[r * LDPAD + c],
                    &W[(size_t)(tiN * 128 + r) * D_SZ + k0 + c]);
    }
    wait_async0();
    __syncthreads();

    v16h af[2], bf[4];
#pragma unroll
    for (int i = 0; i < 2; ++i) af[i] = load_fragA(&As[(wrg + 16 * i) * LDPAD], LDPAD);
#pragma unroll
    for (int j = 0; j < 4; ++j) bf[j] = load_fragB(&Bs[(wcg + 16 * j) * LDPAD], LDPAD);
#pragma unroll
    for (int i = 0; i < 2; ++i)
#pragma unroll
      for (int j = 0; j < 4; ++j)
        acc[i][j] = wmma16(af[i], bf[j], acc[i][j]);
    __syncthreads();
  }

#pragma unroll
  for (int i = 0; i < 2; ++i)
#pragma unroll
    for (int j = 0; j < 4; ++j) {
      const int colLocal = wcg + 16 * j + (lane & 15);
      const int N = tiN * 128 + colLocal;
      const float bv = bias[N];
#pragma unroll
      for (int r = 0; r < 8; ++r) {
        const int rowLocal = wrg + 16 * i + 8 * (lane >> 4) + r;
        const int gRow = tiM * 128 + rowLocal;
        out[(size_t)gRow * D_SZ + N] = acc[i][j][r] + bv;
      }
    }
}

// ---------------------------------------------------------------------------
// Launch
// ---------------------------------------------------------------------------
extern "C" void kernel_launch(void* const* d_in, const int* in_sizes, int n_in,
                              void* d_out, int out_size, void* d_ws, size_t ws_size,
                              hipStream_t stream) {
  const float* query = (const float*)d_in[0];   // [4,2048,1024]
  const float* w_in  = (const float*)d_in[1];   // [3072,1024]
  const float* b_in  = (const float*)d_in[2];   // [3072]
  const float* w_out = (const float*)d_in[3];   // [1024,1024]
  const float* b_out = (const float*)d_in[4];   // [1024]
  float* out = (float*)d_out;                   // [4,2048,1024]

  const size_t nQKV = (size_t)B_SZ * H_SZ * T_SZ * HD_SZ;   // 8,388,608
  _Float16* win_h  = (_Float16*)d_ws;                       // 3,145,728
  _Float16* wout_h = win_h + (size_t)3072 * 1024;           // 1,048,576
  _Float16* Qf = wout_h + (size_t)1024 * 1024;
  _Float16* Kf = Qf + nQKV;
  _Float16* Vt = Kf + nQKV;
  _Float16* Ob = Vt + nQKV;                                 // [8192,1024]

  cvt_f32_f16<<<(3145728 / 4 + 255) / 256, 256, 0, stream>>>(w_in, win_h, 3145728);
  cvt_f32_f16<<<(1048576 / 4 + 255) / 256, 256, 0, stream>>>(w_out, wout_h, 1048576);

  gemm_qkv_rope<<<dim3(24, 64), 256, 0, stream>>>(query, win_h, b_in, Qf, Kf, Vt);
  attn_kernel<<<dim3(64, 8), 256, 0, stream>>>(Qf, Kf, Vt, Ob);
  gemm_out<<<dim3(8, 64), 256, 0, stream>>>(Ob, wout_h, b_out, out);
}
